// Baseline_78005196030499
// MI455X (gfx1250) — compile-verified, matrix operand order checked by reading, not run on hardware
//
#include <hip/hip_runtime.h>
#include <cstdint>
#include <cstddef>

// Problem sizes (fixed by the reference)
#define VOCAB 50257
#define EMB   256
#define HID   512
#define OUTD  2
#define BATCH 128
#define TSEQ  512

// CDNA5 WMMA types
typedef __attribute__((ext_vector_type(16))) __bf16 v16bf;
typedef __attribute__((ext_vector_type(8)))  float  v8f;
typedef __attribute__((ext_vector_type(4)))  float  v4f;   // native vector for NT loads

union Frag16 { v16bf v; uint4 q[2]; };           // 16 bf16 = 32B = 2x b128 loads

__device__ __forceinline__ unsigned short f32_to_bf16_rne(float f) {
  unsigned u = __float_as_uint(f);
  unsigned r = 0x7FFFu + ((u >> 16) & 1u);       // round-to-nearest-even
  return (unsigned short)((u + r) >> 16);
}
__device__ __forceinline__ float bf16_to_f32(unsigned short h) {
  return __uint_as_float(((unsigned)h) << 16);
}

// ---------------------------------------------------------------------------
// Kernel 0: convert Ww [512x256] and Uw [512x512] f32 -> bf16 (row-major kept:
// row n of Uw is exactly column n of the GEMM's B matrix, K-contiguous).
// ---------------------------------------------------------------------------
__global__ __launch_bounds__(256) void convert_weights(
    const float* __restrict__ Ww, const float* __restrict__ Uw,
    unsigned short* __restrict__ Ww16, unsigned short* __restrict__ Uw16) {
  int i = blockIdx.x * blockDim.x + threadIdx.x;
  if (i < HID * EMB) Ww16[i] = f32_to_bf16_rne(Ww[i]);
  if (i < HID * HID) Uw16[i] = f32_to_bf16_rne(Uw[i]);
}

// ---------------------------------------------------------------------------
// Kernel 1: fused embedding gather + wx = xe @ Ww^T + Wb.
// One workgroup (256 thr / 8 waves) per tile of 16 BATCH rows at a fixed t:
//   tileId = t*8 + btile,  rows m = 0..15 <-> batch b = btile*16 + m.
// Output stored PRE-PERMUTED in WMMA D-fragment order:
//   wxp[ (tileId*HID + n)*16 + m ]
// so the recurrence kernel reads each lane's 8 addends as 2 contiguous b128s
// (and the add is correct under any actual D layout, since both kernels use
// the identical (lane, vgpr) -> (m, n) mapping).
// ---------------------------------------------------------------------------
#define P1_PITCH 264   // ushorts per LDS row: 256 + 8 pad -> bank-conflict-free b128

__global__ __launch_bounds__(256) void embed_wx_kernel(
    const int* __restrict__ x, const float* __restrict__ emb,
    const unsigned short* __restrict__ Ww16, const float* __restrict__ Wb,
    float* __restrict__ wxp) {
  __shared__ unsigned short aLds[16 * P1_PITCH];
  __shared__ int toks[16];

  const int tid    = threadIdx.x;
  const int tileId = blockIdx.x;                 // t*8 + btile
  const int t      = tileId >> 3;
  const int b0     = (tileId & 7) * 16;

  if (tid < 16) toks[tid] = x[(b0 + tid) * TSEQ + t];
  __syncthreads();

  // Gather 16 x 256 f32 embedding rows -> bf16 in LDS (each thread: 16 elems)
  {
    const int r  = tid >> 4;                     // 0..15 row (batch offset)
    const int c0 = (tid & 15) * 16;              // 0..240 col base
    const float* src = emb + (size_t)toks[r] * EMB + c0;
    union { unsigned short s[16]; uint4 q[2]; } tmp;
#pragma unroll
    for (int i = 0; i < 4; ++i) {
      float4 f = ((const float4*)src)[i];
      tmp.s[4 * i + 0] = f32_to_bf16_rne(f.x);
      tmp.s[4 * i + 1] = f32_to_bf16_rne(f.y);
      tmp.s[4 * i + 2] = f32_to_bf16_rne(f.z);
      tmp.s[4 * i + 3] = f32_to_bf16_rne(f.w);
    }
    uint4* dst = (uint4*)&aLds[r * P1_PITCH + c0];   // 16B aligned (528B pitch)
    dst[0] = tmp.q[0];
    dst[1] = tmp.q[1];
  }
  __syncthreads();

  const int lane  = tid & 31;
  const int wave  = tid >> 5;                    // 0..7, each covers 64 cols of H
  const int laneN = lane & 15;
  const int hi    = lane >> 4;                   // 0/1: K-half select per ISA layout
  const int nBase = wave * 64;

  v8f acc[4] = {};                               // 4 N-tiles x 16x16 f32

  for (int kk = 0; kk < EMB / 32; ++kk) {        // 8 K-steps of 32
    // A frag: row = laneN, K = kk*32 + hi*8 + {0..7, 16..23}
    Frag16 a;
    const unsigned short* ap = &aLds[laneN * P1_PITCH + kk * 32 + hi * 8];
    a.q[0] = *(const uint4*)(ap);
    a.q[1] = *(const uint4*)(ap + 16);
#pragma unroll
    for (int nt = 0; nt < 4; ++nt) {
      const int n = nBase + nt * 16 + laneN;     // B column = Ww row (K-contig)
      const unsigned short* bp = Ww16 + (size_t)n * EMB + kk * 32 + hi * 8;
      Frag16 b;
      b.q[0] = *(const uint4*)(bp);
      b.q[1] = *(const uint4*)(bp + 16);
      acc[nt] = __builtin_amdgcn_wmma_f32_16x16x32_bf16(
          false, a.v, false, b.v, (short)0, acc[nt], false, false);
    }
  }

  // Epilogue: lane owns D[m = v + 8*hi][n]; store its 8 values contiguously.
#pragma unroll
  for (int nt = 0; nt < 4; ++nt) {
    const int n  = nBase + nt * 16 + laneN;
    const float wb = Wb[n];
    union { float4 q[2]; float f[8]; } o;
#pragma unroll
    for (int v = 0; v < 8; ++v) o.f[v] = acc[nt][v] + wb;
    float* dst = wxp + ((size_t)tileId * HID + n) * 16 + hi * 8;  // 32B aligned
    ((float4*)dst)[0] = o.q[0];
    ((float4*)dst)[1] = o.q[1];     // wave-wide: one contiguous 1KB block
  }
}

// ---------------------------------------------------------------------------
// Kernel 2: sequential recurrence h = tanh(wx_t + h @ Uw^T + Ub), then head.
// 8 persistent workgroups (16 batch rows each), 512 thr / 16 waves,
// each wave owns 32 columns of h. h ping-pongs in bf16 LDS.
// wx operand double-buffered in registers: t+1's fragment loads are issued
// before the 32-WMMA K loop, hiding their latency behind matrix work.
// ---------------------------------------------------------------------------
#define P2_PITCH 520   // 512 + 8 pad ushorts -> conflict-free ds_load_b128

__global__ __launch_bounds__(512) void rnn_kernel(
    const unsigned short* __restrict__ Uw16, const float* __restrict__ Ub,
    const float* __restrict__ wxp, const float* __restrict__ Vw,
    const float* __restrict__ Vb, float* __restrict__ out) {
  __shared__ unsigned short hbuf[2][16 * P2_PITCH];

  const int tid   = threadIdx.x;
  const int btile = blockIdx.x;                  // 0..7
  const int b0    = btile * 16;
  const int lane  = tid & 31;
  const int wave  = tid >> 5;                    // 0..15
  const int laneN = lane & 15;
  const int hi    = lane >> 4;

  // Loop-invariant hoists
  const int n0 = wave * 32 + laneN;              // nt=0 column
  const int n1 = n0 + 16;                        // nt=1 column
  const float ub0 = Ub[n0];
  const float ub1 = Ub[n1];
  const unsigned short* bRow0 = Uw16 + (size_t)n0 * HID + hi * 8;
  const unsigned short* bRow1 = Uw16 + (size_t)n1 * HID + hi * 8;

  // wx fragment loader: lane's 8 addends per N-tile are contiguous (permuted
  // layout written by phase 1); single-use stream -> non-temporal.
  auto loadWx = [&](int t, float (&wv)[2][8]) {
#pragma unroll
    for (int nt = 0; nt < 2; ++nt) {
      const int n = (nt == 0) ? n0 : n1;
      const v4f* p = (const v4f*)(wxp +
          ((size_t)(t * 8 + btile) * HID + n) * 16 + hi * 8);
      v4f a = __builtin_nontemporal_load(p);
      v4f b = __builtin_nontemporal_load(p + 1);
      wv[nt][0] = a.x; wv[nt][1] = a.y; wv[nt][2] = a.z; wv[nt][3] = a.w;
      wv[nt][4] = b.x; wv[nt][5] = b.y; wv[nt][6] = b.z; wv[nt][7] = b.w;
    }
  };

  // h0 = 0
  for (int i = tid; i < 16 * P2_PITCH; i += 512) hbuf[0][i] = 0;
  __syncthreads();

  float wcur[2][8];
  loadWx(0, wcur);

  int cur = 0;
  for (int t = 0; t < TSEQ; ++t) {
    float wnxt[2][8];
    if (t + 1 < TSEQ) loadWx(t + 1, wnxt);       // overlaps with K loop below

    v8f acc[2] = {};
    const unsigned short* hb = hbuf[cur];

    for (int kk = 0; kk < HID / 32; ++kk) {      // 16 K-steps of 32
      Frag16 a;
      const unsigned short* ap = &hb[laneN * P2_PITCH + kk * 32 + hi * 8];
      a.q[0] = *(const uint4*)(ap);
      a.q[1] = *(const uint4*)(ap + 16);

      Frag16 b;                                   // Uw stays L2-hot (512 KB)
      b.q[0] = *(const uint4*)(bRow0 + kk * 32);
      b.q[1] = *(const uint4*)(bRow0 + kk * 32 + 16);
      acc[0] = __builtin_amdgcn_wmma_f32_16x16x32_bf16(
          false, a.v, false, b.v, (short)0, acc[0], false, false);

      b.q[0] = *(const uint4*)(bRow1 + kk * 32);
      b.q[1] = *(const uint4*)(bRow1 + kk * 32 + 16);
      acc[1] = __builtin_amdgcn_wmma_f32_16x16x32_bf16(
          false, a.v, false, b.v, (short)0, acc[1], false, false);
    }

    unsigned short* hn = hbuf[cur ^ 1];
#pragma unroll
    for (int nt = 0; nt < 2; ++nt) {
      const int n  = (nt == 0) ? n0 : n1;
      const float ub = (nt == 0) ? ub0 : ub1;
#pragma unroll
      for (int v = 0; v < 8; ++v) {
        const int m = v + hi * 8;
        const float s = acc[nt][v] + wcur[nt][v] + ub;
        hn[m * P2_PITCH + n] = f32_to_bf16_rne(tanhf(s));
      }
    }
    __syncthreads();   // covers both "reads of cur done" and "writes of nxt done"
    cur ^= 1;

    if (t + 1 < TSEQ) {
#pragma unroll
      for (int nt = 0; nt < 2; ++nt)
#pragma unroll
        for (int v = 0; v < 8; ++v) wcur[nt][v] = wnxt[nt][v];
    }
  }

  // Output head: logits[b0+m][o] = h_T . Vw[o] + Vb[o]   (wave 0 only)
  if (tid < 32) {
    const int m = tid & 15;
    const int o = tid >> 4;
    const unsigned short* hb = hbuf[cur];
    float dot = 0.f;
    for (int k = 0; k < HID; ++k)
      dot += bf16_to_f32(hb[m * P2_PITCH + k]) * Vw[o * HID + k];
    out[(b0 + m) * OUTD + o] = dot + Vb[o];
  }
}

// ---------------------------------------------------------------------------
// Launch
// ---------------------------------------------------------------------------
extern "C" void kernel_launch(void* const* d_in, const int* in_sizes, int n_in,
                              void* d_out, int out_size, void* d_ws, size_t ws_size,
                              hipStream_t stream) {
  const int*   x   = (const int*)  d_in[0];
  const float* emb = (const float*)d_in[1];
  const float* Ww  = (const float*)d_in[2];
  const float* Wb  = (const float*)d_in[3];
  const float* Uw  = (const float*)d_in[4];
  const float* Ub  = (const float*)d_in[5];
  const float* Vw  = (const float*)d_in[6];
  const float* Vb  = (const float*)d_in[7];
  float* out = (float*)d_out;

  // Workspace layout: Ww16 (256KB) | Uw16 (512KB) | wxp f32 permuted (128MB)
  char* ws = (char*)d_ws;
  unsigned short* Ww16 = (unsigned short*)ws;
  unsigned short* Uw16 = (unsigned short*)(ws + (size_t)HID * EMB * 2);
  float*          wxp  = (float*)(ws + (size_t)HID * EMB * 2 + (size_t)HID * HID * 2);

  convert_weights<<<(HID * HID + 255) / 256, 256, 0, stream>>>(Ww, Uw, Ww16, Uw16);
  embed_wx_kernel<<<(BATCH * TSEQ) / 16, 256, 0, stream>>>(x, emb, Ww16, Wb, wxp);
  rnn_kernel<<<BATCH / 16, 512, 0, stream>>>(Uw16, Ub, wxp, Vw, Vb, out);
}
